// Transformer_4913442586973
// MI455X (gfx1250) — matrix operand order repaired and checked by reading, not compiled, not run on hardware
//
#include <hip/hip_runtime.h>
#include <hip/hip_bf16.h>
#include <math.h>

// ---------------- problem constants ----------------
#define BB   4
#define SS   4096
#define DD   512
#define HH   8
#define LL   6
#define DFFC 2048
#define VSC  69
#define VTC  69
#define DKC  64
#define NFC  7
#define FEATC 34
#define NCH  64      // chunks along S for the performer scan
#define CHL  64      // chunk length (SS / NCH)
#define EPSF 1e-6f

// ---------------- WMMA / vector types ----------------
typedef __attribute__((ext_vector_type(16))) __bf16 bf16x16;
typedef __attribute__((ext_vector_type(8)))  float  f32x8;
typedef unsigned int uint32x4v __attribute__((ext_vector_type(4)));
typedef int          int32x4v  __attribute__((ext_vector_type(4)));
typedef int          int32x8v  __attribute__((ext_vector_type(8)));

union FragBF { unsigned int u[8]; bf16x16 v; };

__device__ __forceinline__ unsigned short f2bf(float f) {
  unsigned int u = __float_as_uint(f);
  unsigned int r = u + 0x7FFFu + ((u >> 16) & 1u);   // round-to-nearest-even
  return (unsigned short)(r >> 16);
}

__device__ __forceinline__ float gelu_f(float x) {
  const float c = 0.7978845608028654f;               // sqrt(2/pi)
  float t = tanhf(c * (x + 0.044715f * x * x * x));
  return 0.5f * x * (1.0f + t);
}

// ---------------- packed f32 -> bf16 conversion (weights) ----------------
__global__ void cvt_bf16_kernel(const float* __restrict__ src,
                                unsigned short* __restrict__ dst, int n4)
{
  int i = blockIdx.x * blockDim.x + threadIdx.x;
  if (i >= n4) return;
  float4 v = reinterpret_cast<const float4*>(src)[i];
  ushort4 o;
  o.x = f2bf(v.x); o.y = f2bf(v.y); o.z = f2bf(v.z); o.w = f2bf(v.w);
  reinterpret_cast<ushort4*>(dst)[i] = o;
}

// ---------------- embedding + positional encoding ----------------
__global__ void embed_kernel(const int* __restrict__ tok,
                             const float* __restrict__ id_embed,
                             const float* __restrict__ feat_w,   // (D, FEAT)
                             const float* __restrict__ feat_t,   // (VS, FEAT)
                             const unsigned char* __restrict__ pmask,
                             float* __restrict__ x)
{
  int idx = blockIdx.x * blockDim.x + threadIdx.x;
  if (idx >= BB * SS * DD) return;
  int d = idx % DD;
  int s = (idx / DD) % SS;
  int b = idx / (DD * SS);
  int t = tok[b * SS + s];
  float val = id_embed[(size_t)t * DD + d];
  if (pmask[t]) {
    float acc = 0.f;
    #pragma unroll
    for (int j = 0; j < FEATC; j++)
      acc += feat_t[(size_t)t * FEATC + j] * feat_w[(size_t)d * FEATC + j];
    val += acc;
  }
  float i2  = (float)(d & ~1);
  float div = expf(-logf(10000.f) * i2 / (float)DD);
  float ang = (float)s * div;
  val += (d & 1) ? cosf(ang) : sinf(ang);
  x[idx] = val;
}

// ---------------- layernorm -> bf16 output (feeds GEMMs only) ----------------
__global__ void ln_kernel(const float* __restrict__ x,
                          const float* __restrict__ alpha,
                          const float* __restrict__ beta,
                          unsigned short* __restrict__ y)
{
  int row = blockIdx.x;                 // B*S rows
  const float* xr = x + (size_t)row * DD;
  int t = threadIdx.x;                  // 128 threads, 4 elems each
  float v[4];
  float s = 0.f;
  #pragma unroll
  for (int i = 0; i < 4; i++) { v[i] = xr[t + 128 * i]; s += v[i]; }
  __shared__ float red[128];
  red[t] = s; __syncthreads();
  for (int o = 64; o > 0; o >>= 1) { if (t < o) red[t] += red[t + o]; __syncthreads(); }
  float mean = red[0] / (float)DD;
  __syncthreads();
  float sq = 0.f;
  #pragma unroll
  for (int i = 0; i < 4; i++) { float d0 = v[i] - mean; sq += d0 * d0; }
  red[t] = sq; __syncthreads();
  for (int o = 64; o > 0; o >>= 1) { if (t < o) red[t] += red[t + o]; __syncthreads(); }
  float var = red[0] / (float)(DD - 1);
  float inv = 1.f / (sqrtf(var) + EPSF);
  unsigned short* yr = y + (size_t)row * DD;
  #pragma unroll
  for (int i = 0; i < 4; i++) {
    int d0 = t + 128 * i;
    yr[d0] = f2bf(alpha[d0] * (v[i] - mean) * inv + beta[d0]);
  }
}

// ---------------- TDM helper: 2D bf16 tile -> LDS with skew padding ----------------
#if __has_builtin(__builtin_amdgcn_tensor_load_to_lds)
#define HAVE_TDM 1
__device__ __forceinline__ void tdm_load_2d_bf16(unsigned lds_off, const void* gaddr,
                                                 int rows, int row_stride_elems)
{
  unsigned long long ga = (unsigned long long)(size_t)gaddr;
  uint32x4v g0;
  g0[0] = 1u;                                               // count=1 (valid user D#)
  g0[1] = lds_off;                                          // lds_addr
  g0[2] = (unsigned)(ga & 0xFFFFFFFFull);                   // global_addr[31:0]
  g0[3] = (unsigned)((ga >> 32) & 0x01FFFFFFull) | 0x80000000u; // addr[56:32], type=2
  int32x8v g1;
  // data_size=1 (2B), pad_enable, pad_interval=3 (16 DWORDs), pad_amount=0 (1 DWORD)
  g1[0] = (int)((1u << 16) | (1u << 20) | (3u << 22));
  g1[1] = (int)(32u << 16);                                 // tensor_dim0 = 32 (lo16)
  g1[2] = (int)(((unsigned)rows & 0xFFFFu) << 16);          // tensor_dim1 = rows (lo16)
  g1[3] = (int)(32u << 16);                                 // tile_dim0 = 32
  g1[4] = rows;                                             // tile_dim1 = rows, tile_dim2 = 0
  g1[5] = row_stride_elems;                                 // tensor_dim0_stride lo32
  g1[6] = 0;
  g1[7] = 0;
  int32x4v z4 = {0, 0, 0, 0};
#if __clang_major__ >= 23
  int32x8v z8 = {0, 0, 0, 0, 0, 0, 0, 0};
  __builtin_amdgcn_tensor_load_to_lds(g0, g1, z4, z4, z8, 0);
#else
  __builtin_amdgcn_tensor_load_to_lds(g0, g1, z4, z4, 0);
#endif
}
#else
#define HAVE_TDM 0
#endif

// ---------------- fragment load + 2x2 WMMA from one LDS stage ----------------
__device__ __forceinline__ void frags_and_wmma(
    const unsigned short (* __restrict__ Asb)[34],
    const unsigned short (* __restrict__ Bsb)[34],
    int wm, int wn, int mr, int kbA, int kbB, f32x8 (&acc)[2][2])
{
  FragBF af[2], bfr[2];
  #pragma unroll
  for (int tm = 0; tm < 2; tm++) {
    const unsigned int* pr = (const unsigned int*)Asb[wm * 32 + tm * 16 + mr];
    #pragma unroll
    for (int vv = 0; vv < 8; vv++) af[tm].u[vv]     = pr[(kbA >> 1) + vv];        // K=kb..kb+15
  }
  #pragma unroll
  for (int tn = 0; tn < 2; tn++) {
    const unsigned int* pr = (const unsigned int*)Bsb[wn * 32 + tn * 16 + mr];
    #pragma unroll
    for (int vv = 0; vv < 8; vv++) bfr[tn].u[vv] = pr[(kbB >> 1) + vv];           // K=kb2..kb2+15
  }
  #pragma unroll
  for (int tm = 0; tm < 2; tm++)
    #pragma unroll
    for (int tn = 0; tn < 2; tn++)
      acc[tm][tn] = __builtin_amdgcn_wmma_f32_16x16x32_bf16(
          false, af[tm].v, false, bfr[tn].v, (short)0, acc[tm][tn], false, false);
}

// ---------------- bf16 WMMA GEMM:  C = act(A @ W^T + bias) + res ----------------
// A: (M,K) bf16 row-major.  W: (N,K) bf16 row-major (so B(k,n)=W[n][k]).
// Block tile 64x128, 8 waves, each wave a 32x32 sub-tile of 2x2 WMMA tiles.
// Full tiles use a double-buffered Tensor-Data-Mover pipeline (DMA overlaps WMMA).
// Output to Cf (f32) or Cb (bf16) - exactly one non-null.
__global__ __launch_bounds__(256) void gemm_bf16_wmma(
    const unsigned short* __restrict__ A, const unsigned short* __restrict__ W,
    const float* __restrict__ bias, const float* __restrict__ res,
    float* __restrict__ Cf, unsigned short* __restrict__ Cb,
    int M, int N, int K, int act)
{
  __shared__ unsigned short As[2][64][34];    // [m][k] bf16, stride 34 (dword aligned, skewed)
  __shared__ unsigned short Bs[2][128][34];   // [n][k] bf16
  (void)M;
  int tid  = threadIdx.x;
  int lane = tid & 31;
  int w    = tid >> 5;
  int wm   = w & 1;            // wave row  (2 along M)
  int wn   = w >> 1;           // wave col  (4 along N)
  int m0   = blockIdx.x * 64;
  int n0   = blockIdx.y * 128;

  f32x8 acc[2][2] = {};

  int mr  = lane & 15;
  int kbA = (lane < 16) ? 0 : 16;   // lane half h: A k = 16h..16h+15 (contiguous order)
  int kbB = (lane < 16) ? 0 : 16;   // lane half h: B k = 16h..16h+15 (same order as A)

  bool fullB = (n0 + 128 <= N);

#if HAVE_TDM
  if (fullB) {
    // ---- double-buffered TDM pipeline ----
    unsigned ldsA[2] = { (unsigned)(unsigned long long)(size_t)&As[0][0][0],
                         (unsigned)(unsigned long long)(size_t)&As[1][0][0] };
    unsigned ldsB[2] = { (unsigned)(unsigned long long)(size_t)&Bs[0][0][0],
                         (unsigned)(unsigned long long)(size_t)&Bs[1][0][0] };
    const unsigned short* gA = A + (size_t)m0 * K;
    const unsigned short* gW = W + (size_t)n0 * K;
    int nk = K >> 5;
    if (w == 0) {
      tdm_load_2d_bf16(ldsA[0], gA, 64, K);
      tdm_load_2d_bf16(ldsB[0], gW, 128, K);
      __builtin_amdgcn_s_wait_tensorcnt(0);
    }
    __syncthreads();
    for (int ki = 0; ki < nk; ki++) {
      int cur = ki & 1, nxt = cur ^ 1;
      bool more = (ki + 1 < nk);
      if (w == 0 && more) {
        int k1 = (ki + 1) << 5;
        tdm_load_2d_bf16(ldsA[nxt], gA + k1, 64, K);   // async: overlaps WMMA below
        tdm_load_2d_bf16(ldsB[nxt], gW + k1, 128, K);
      }
      frags_and_wmma(As[cur], Bs[cur], wm, wn, mr, kbA, kbB, acc);
      if (w == 0 && more) __builtin_amdgcn_s_wait_tensorcnt(0);
      __syncthreads();
    }
  } else
#endif
  {
    // ---- manual path (partial N tiles / no TDM): 128-bit loads, single buffer ----
    for (int k0 = 0; k0 < K; k0 += 32) {
      {
        int rowA = tid >> 2;                 // 0..63
        int colA = (tid & 3) * 8;            // 0,8,16,24
        uint4 va = *reinterpret_cast<const uint4*>(A + (size_t)(m0 + rowA) * K + k0 + colA);
        unsigned int* dst = (unsigned int*)&As[0][rowA][colA];
        dst[0] = va.x; dst[1] = va.y; dst[2] = va.z; dst[3] = va.w;
      }
      #pragma unroll
      for (int i = 0; i < 2; i++) {
        int e = tid + 256 * i;
        int rowB = e >> 2;                   // 0..127
        int colB = (e & 3) * 8;
        int n = n0 + rowB;
        uint4 vb = make_uint4(0u, 0u, 0u, 0u);
        if (n < N)
          vb = *reinterpret_cast<const uint4*>(W + (size_t)n * K + k0 + colB);
        unsigned int* dst = (unsigned int*)&Bs[0][rowB][colB];
        dst[0] = vb.x; dst[1] = vb.y; dst[2] = vb.z; dst[3] = vb.w;
      }
      __syncthreads();
      frags_and_wmma(As[0], Bs[0], wm, wn, mr, kbA, kbB, acc);
      __syncthreads();
    }
  }

  // epilogue: 32-bit C/D layout. lane<16 -> rows 0..7 ; lane>=16 -> rows 8..15
  int rbase = (lane < 16) ? 0 : 8;
  #pragma unroll
  for (int tm = 0; tm < 2; tm++)
    #pragma unroll
    for (int tn = 0; tn < 2; tn++) {
      int col = n0 + wn * 32 + tn * 16 + mr;
      if (col >= N) continue;
      #pragma unroll
      for (int r = 0; r < 8; r++) {
        int row = m0 + wm * 32 + tm * 16 + rbase + r;
        float val = acc[tm][tn][r];
        if (bias) val += bias[col];
        if (act)  val = gelu_f(val);
        size_t off = (size_t)row * N + col;
        if (res)  val += res[off];
        if (Cb) Cb[off] = f2bf(val);
        else    Cf[off] = val;
      }
    }
}

// ---------------- performer feature maps (qp, kp normalized over NF) ----------------
__global__ void feat_kernel(const float* __restrict__ q, const float* __restrict__ k,
                            const float* __restrict__ omega,     // (NF, DK)
                            float* __restrict__ qp, float* __restrict__ kp)
{
  int idx = blockIdx.x * blockDim.x + threadIdx.x;   // (b,s,h)
  if (idx >= BB * SS * HH) return;
  int h  = idx % HH;
  int bs = idx / HH;
  const float* qr = q + (size_t)bs * DD + h * DKC;
  const float* kr = k + (size_t)bs * DD + h * DKC;
  float eq[NFC], ek[NFC], sq = 0.f, sk = 0.f;
  #pragma unroll
  for (int f = 0; f < NFC; f++) {
    float dq = 0.f, dk = 0.f;
    const float* om = omega + f * DKC;
    for (int d0 = 0; d0 < DKC; d0++) { dq += qr[d0] * om[d0]; dk += kr[d0] * om[d0]; }
    eq[f] = expf(-0.5f * dq * dq);
    ek[f] = expf(-0.5f * dk * dk);
    sq += eq[f]; sk += ek[f];
  }
  float iq = 1.f / (sq + EPSF), ik = 1.f / (sk + EPSF);
  float* qpr = qp + (size_t)idx * NFC;
  float* kpr = kp + (size_t)idx * NFC;
  #pragma unroll
  for (int f = 0; f < NFC; f++) { qpr[f] = eq[f] * iq; kpr[f] = ek[f] * ik; }
}

// ---------------- blocked prefix scan over S: phase 1 (per-chunk sums) ----------------
__global__ void scan_pass1(const float* __restrict__ kp, const float* __restrict__ v,
                           float* __restrict__ ckv, float* __restrict__ ck)
{
  int blk = blockIdx.x;                  // bh*NCH + c
  int c  = blk % NCH;
  int bh = blk / NCH;
  int b = bh / HH, h = bh % HH;
  int d = threadIdx.x;                   // 64 threads = DK
  float acc[NFC];
  #pragma unroll
  for (int f = 0; f < NFC; f++) acc[f] = 0.f;
  float accK = 0.f;
  int s0 = c * CHL;
  for (int i = 0; i < CHL; i++) {
    size_t pos = (size_t)(b * SS + s0 + i);
    float vd = v[pos * DD + h * DKC + d];
    const float* kpr = kp + (pos * HH + h) * NFC;
    #pragma unroll
    for (int f = 0; f < NFC; f++) acc[f] += kpr[f] * vd;
    if (d < NFC) accK += kpr[d];
  }
  float* o = ckv + (size_t)blk * NFC * DKC;
  #pragma unroll
  for (int f = 0; f < NFC; f++) o[f * DKC + d] = acc[f];
  if (d < NFC) ck[(size_t)blk * NFC + d] = accK;
}

// ---------------- phase 2: exclusive scan over chunks (in place) ----------------
__global__ void scan_chunks(float* __restrict__ ckv, float* __restrict__ ck)
{
  int bh = blockIdx.x;                   // B*H
  int t  = threadIdx.x;                  // 512 threads; 448 (f,d) + 7 for K
  if (t < NFC * DKC) {
    float run = 0.f;
    for (int c = 0; c < NCH; c++) {
      size_t ix = (size_t)(bh * NCH + c) * NFC * DKC + t;
      float tv = ckv[ix]; ckv[ix] = run; run += tv;
    }
  } else if (t < NFC * DKC + NFC) {
    int f = t - NFC * DKC;
    float run = 0.f;
    for (int c = 0; c < NCH; c++) {
      size_t ix = (size_t)(bh * NCH + c) * NFC + f;
      float tv = ck[ix]; ck[ix] = run; run += tv;
    }
  }
}

// ---------------- phase 3: in-chunk inclusive replay + num/den, bf16 out ----------------
__global__ void scan_pass2(const float* __restrict__ qp, const float* __restrict__ kp,
                           const float* __restrict__ v,
                           const float* __restrict__ ckv, const float* __restrict__ ck,
                           unsigned short* __restrict__ attnb)
{
  int blk = blockIdx.x;
  int c  = blk % NCH;
  int bh = blk / NCH;
  int b = bh / HH, h = bh % HH;
  int d = threadIdx.x;                   // 64 threads = DK
  float acc[NFC], Kc[NFC];
  const float* pv = ckv + (size_t)blk * NFC * DKC;
  const float* pk = ck  + (size_t)blk * NFC;
  #pragma unroll
  for (int f = 0; f < NFC; f++) { acc[f] = pv[f * DKC + d]; Kc[f] = pk[f]; }
  int s0 = c * CHL;
  for (int i = 0; i < CHL; i++) {
    size_t pos = (size_t)(b * SS + s0 + i);
    float vd = v[pos * DD + h * DKC + d];
    const float* kpr = kp + (pos * HH + h) * NFC;
    const float* qpr = qp + (pos * HH + h) * NFC;
    float num = 0.f, den = 0.f;
    #pragma unroll
    for (int f = 0; f < NFC; f++) {
      acc[f] += kpr[f] * vd;
      Kc[f]  += kpr[f];
      num += qpr[f] * acc[f];
      den += qpr[f] * Kc[f];
    }
    attnb[pos * DD + h * DKC + d] = f2bf(num / (den + EPSF));
  }
}

// ---------------- driver ----------------
extern "C" void kernel_launch(void* const* d_in, const int* in_sizes, int n_in,
                              void* d_out, int out_size, void* d_ws, size_t ws_size,
                              hipStream_t stream)
{
  (void)in_sizes; (void)n_in; (void)out_size; (void)ws_size;
  const int*   tok      = (const int*)d_in[0];
  const float* id_embed = (const float*)d_in[1];
  const float* feat_w   = (const float*)d_in[2];
  const float* feat_t   = (const float*)d_in[3];
  const unsigned char* pmask = (const unsigned char*)d_in[4];
  const float* wq    = (const float*)d_in[5];
  const float* wk    = (const float*)d_in[6];
  const float* wv    = (const float*)d_in[7];
  const float* wo    = (const float*)d_in[8];
  const float* omega = (const float*)d_in[9];
  const float* ln1a  = (const float*)d_in[10];
  const float* ln1b  = (const float*)d_in[11];
  const float* ln2a  = (const float*)d_in[12];
  const float* ln2b  = (const float*)d_in[13];
  const float* w1    = (const float*)d_in[14];
  const float* b1    = (const float*)d_in[15];
  const float* w2    = (const float*)d_in[16];
  const float* b2    = (const float*)d_in[17];
  const float* fina  = (const float*)d_in[18];
  const float* finb  = (const float*)d_in[19];
  const float* pw    = (const float*)d_in[20];
  const float* pb    = (const float*)d_in[21];
  float* out = (float*)d_out;

  const int M = BB * SS;                         // 16384

  // -------- workspace layout --------
  const size_t XN   = (size_t)BB * SS * DD;      // 8.4M
  const size_t QPN  = (size_t)BB * SS * HH * NFC;
  const size_t CKVN = (size_t)BB * HH * NCH * NFC * DKC;
  const size_t CKN  = (size_t)BB * HH * NCH * NFC;
  const size_t H1N  = (size_t)BB * SS * DFFC;    // 33.5M
  const size_t LDD2 = (size_t)LL * DD * DD;      // per-proj weight count
  const size_t LFF  = (size_t)LL * DFFC * DD;
  const size_t PWN  = (size_t)VTC * DD;

  float* ws = (float*)d_ws;
  float* x   = ws;
  float* q   = ws + 1 * XN;
  float* k   = ws + 2 * XN;
  float* v   = ws + 3 * XN;
  float* qp  = ws + 4 * XN;
  float* kp  = qp + QPN;
  float* ckv = kp + QPN;
  float* ck  = ckv + CKVN;
  // bf16 region (offsets all 16B-aligned by construction)
  unsigned short* ub   = (unsigned short*)(ck + CKN);
  unsigned short* xnb  = ub;                      // (M, D)
  unsigned short* attnb= xnb + XN;                // (M, D)
  unsigned short* h1b  = attnb + XN;              // (M, DFF)
  unsigned short* wqb  = h1b + H1N;
  unsigned short* wkb  = wqb + LDD2;
  unsigned short* wvb  = wkb + LDD2;
  unsigned short* wob  = wvb + LDD2;
  unsigned short* w1b  = wob + LDD2;
  unsigned short* w2b  = w1b + LFF;
  unsigned short* pwb  = w2b + LFF;

  // -------- one-time (per launch) weight conversion to bf16 --------
  {
    struct { const float* s; unsigned short* d; size_t n; } cv[7] = {
      { wq, wqb, LDD2 }, { wk, wkb, LDD2 }, { wv, wvb, LDD2 }, { wo, wob, LDD2 },
      { w1, w1b, LFF },  { w2, w2b, LFF },  { pw, pwb, PWN },
    };
    for (int i = 0; i < 7; i++) {
      int n4 = (int)(cv[i].n / 4);
      cvt_bf16_kernel<<<(n4 + 255) / 256, 256, 0, stream>>>(cv[i].s, cv[i].d, n4);
    }
  }

  embed_kernel<<<(BB * SS * DD + 255) / 256, 256, 0, stream>>>(
      tok, id_embed, feat_w, feat_t, pmask, x);

  dim3 gD(M / 64, (DD + 127) / 128);
  dim3 gF(M / 64, (DFFC + 127) / 128);

  for (int l = 0; l < LL; l++) {
    ln_kernel<<<M, 128, 0, stream>>>(x, ln1a + l * DD, ln1b + l * DD, xnb);
    gemm_bf16_wmma<<<gD, 256, 0, stream>>>(xnb, wqb + (size_t)DD * DD * l, nullptr, nullptr, q, nullptr, M, DD, DD, 0);
    gemm_bf16_wmma<<<gD, 256, 0, stream>>>(xnb, wkb + (size_t)DD * DD * l, nullptr, nullptr, k, nullptr, M, DD, DD, 0);
    gemm_bf16_wmma<<<gD, 256, 0, stream>>>(xnb, wvb + (size_t)DD * DD * l, nullptr, nullptr, v, nullptr, M, DD, DD, 0);
    feat_kernel<<<(BB * SS * HH + 255) / 256, 256, 0, stream>>>(
        q, k, omega + (size_t)l * NFC * DKC, qp, kp);
    scan_pass1<<<BB * HH * NCH, DKC, 0, stream>>>(kp, v, ckv, ck);
    scan_chunks<<<BB * HH, 512, 0, stream>>>(ckv, ck);
    scan_pass2<<<BB * HH * NCH, DKC, 0, stream>>>(qp, kp, v, ckv, ck, attnb);
    gemm_bf16_wmma<<<gD, 256, 0, stream>>>(attnb, wob + (size_t)DD * DD * l, nullptr, x, x, nullptr, M, DD, DD, 0);
    ln_kernel<<<M, 128, 0, stream>>>(x, ln2a + l * DD, ln2b + l * DD, xnb);
    gemm_bf16_wmma<<<gF, 256, 0, stream>>>(xnb, w1b + (size_t)DFFC * DD * l, b1 + (size_t)l * DFFC, nullptr, nullptr, h1b, M, DFFC, DD, 1);
    gemm_bf16_wmma<<<gD, 256, 0, stream>>>(h1b, w2b + (size_t)DFFC * DD * l, b2 + (size_t)l * DD, x, x, nullptr, M, DD, DFFC, 0);
  }

  ln_kernel<<<M, 128, 0, stream>>>(x, fina, finb, xnb);
  dim3 gP(M / 64, (VTC + 127) / 128);
  gemm_bf16_wmma<<<gP, 256, 0, stream>>>(xnb, pwb, pb, nullptr, out, nullptr, M, VTC, DD, 0);
}
